// Encoder_transformer_17978733101772
// MI455X (gfx1250) — matrix operand order repaired and checked by reading, not compiled, not run on hardware
//
#include <hip/hip_runtime.h>
#include <hip/hip_bf16.h>
#include <math.h>

// ---------------------------------------------------------------------------
// Problem constants (from reference)
// ---------------------------------------------------------------------------
#define NN    10000
#define EE    160000
#define INC   256
#define MIDC  128
#define OUTC  128
#define HH    8
#define EDC   64
#define HC    1024          // H * 128
#define GRP   16
#define EPSV  1e-5f

typedef __attribute__((ext_vector_type(16))) __bf16 v16bf;
typedef __attribute__((ext_vector_type(8)))  __bf16 v8bf;
typedef __attribute__((ext_vector_type(8)))  float  v8f;

// ---------------------------------------------------------------------------
// gfx1250 async global->LDS staging via inline asm (CDNA5 ISA):
//   GLOBAL_LOAD_ASYNC_TO_LDS_B128  vdst(LDS byte addr), vaddr(64b global), off
// tracked by ASYNCcnt; completed with s_wait_asynccnt.
// ---------------------------------------------------------------------------
#define USE_ASYNC_LDS 1

typedef __attribute__((address_space(3))) char* as3_cp;

__device__ __forceinline__ void async_copy_b128(const void* g, void* l) {
#if USE_ASYNC_LDS
    unsigned           lofs = (unsigned)(size_t)(as3_cp)l;       // LDS byte offset
    unsigned long long ga   = (unsigned long long)(size_t)g;     // global address
    asm volatile("global_load_async_to_lds_b128 %0, %1, off"
                 :: "v"(lofs), "v"(ga) : "memory");
#else
    *(v8bf*)l = *(const v8bf*)g;
#endif
}

__device__ __forceinline__ void wait_async_stage() {
#if USE_ASYNC_LDS
#if __has_builtin(__builtin_amdgcn_s_wait_asynccnt)
    __builtin_amdgcn_s_wait_asynccnt(0);
#else
    asm volatile("s_wait_asynccnt 0x0" ::: "memory");
#endif
#endif
}

// ---------------------------------------------------------------------------
// small helpers
// ---------------------------------------------------------------------------
__device__ __forceinline__ __bf16 f2bf(float f) {
    unsigned u = __float_as_uint(f);
    unsigned r = u + 0x7FFFu + ((u >> 16) & 1u);   // round-to-nearest-even
    unsigned short h = (unsigned short)(r >> 16);
    return __builtin_bit_cast(__bf16, h);
}
__device__ __forceinline__ float bf2f(__bf16 b) {
    unsigned short h = __builtin_bit_cast(unsigned short, b);
    return __uint_as_float(((unsigned)h) << 16);
}
// order-preserving float <-> uint mapping (for atomicMax-based segment max)
__device__ __forceinline__ unsigned flipf(float f) {
    unsigned u = __float_as_uint(f);
    return u ^ ((((int)u) >> 31) | 0x80000000u);
}
__device__ __forceinline__ float unflipf(unsigned v) {
    unsigned mask = (v & 0x80000000u) ? 0x80000000u : 0xFFFFFFFFu;
    return __uint_as_float(v ^ mask);
}

// ---------------------------------------------------------------------------
// elementwise f32 -> bf16
// ---------------------------------------------------------------------------
__global__ __launch_bounds__(256)
void cvt_f32_bf16(const float* __restrict__ in, __bf16* __restrict__ out, int n)
{
    int i = blockIdx.x * blockDim.x + threadIdx.x;
    if (i < n) out[i] = f2bf(in[i]);
}

// W (k x m, row-major f32)  ->  Wt (m x k, row-major bf16)
__global__ __launch_bounds__(256)
void cvt_transpose_f32_bf16(const float* __restrict__ W, __bf16* __restrict__ Wt,
                            int k, int m)
{
    int i = blockIdx.x * blockDim.x + threadIdx.x;
    if (i >= k * m) return;
    int row = i / m;            // K index
    int col = i - row * m;      // M index
    Wt[(size_t)col * k + row] = f2bf(W[i]);
}

// ---------------------------------------------------------------------------
// WMMA GEMM on pre-converted bf16:  Y(n x m) = Xb(n x K) @ Wt(m x K)^T (+bias)
// Block = 4 waves covering 64 rows x 64 cols. The 64-col strip of Wt is shared
// by all waves -> staged once into LDS via async global->LDS, padded to a
// 528-byte row stride so B fragment reads are bank-conflict-free.
// A fragment (16x32 bf16): lanes 0-15 row=l16, K = {0..7,16..23}; lanes 16-31 K+8.
// B fragment (32x16 bf16): lane = column, lanes 0-15 K = {0..7,16..23}; lanes 16-31 K+8.
// ---------------------------------------------------------------------------
template <int K, bool BF16OUT>
__global__ __launch_bounds__(128)
void wmma_gemm_bf16(const __bf16* __restrict__ Xb, const __bf16* __restrict__ Wt,
                    const float* __restrict__ bias, void* __restrict__ Yv,
                    int n, int m)
{
    constexpr int KP = K + 8;                       // pad: 16B, breaks 512B stride
    __shared__ __bf16 ldsB[64 * KP];

    const int tid     = threadIdx.x;
    const int wave    = tid >> 5;
    const int lane    = tid & 31;
    const int half    = lane >> 4;                  // 0 or 1
    const int l16     = lane & 15;
    const int rowBase = (blockIdx.y * 4 + wave) * 16;
    const int colBase = blockIdx.x * 64;

    // ---- cooperative stage of Wt[colBase..colBase+63][0..K) into LDS ----
    constexpr int CHUNKS = 64 * K / 8;              // 16-byte chunks
    constexpr int CPC    = K / 8;                   // chunks per column
#pragma unroll
    for (int c = tid; c < CHUNKS; c += 128) {
        const int col = c / CPC;
        const int k8  = (c - col * CPC) * 8;
        async_copy_b128(Wt + (size_t)(colBase + col) * K + k8,
                        &ldsB[col * KP + k8]);
    }
    wait_async_stage();
    __syncthreads();

    if (rowBase >= n) return;                       // wave-uniform, EXEC all-ones

    const __bf16* ap = Xb + (size_t)(rowBase + l16) * K + half * 8;

    v8f acc0 = {}, acc1 = {}, acc2 = {}, acc3 = {};
#pragma unroll
    for (int k0 = 0; k0 < K; k0 += 32) {
        if (k0 + 32 < K)                            // gfx1250 global_prefetch_b8
            __builtin_prefetch(ap + k0 + 32, 0, 0);
        v8bf lo = *(const v8bf*)(ap + k0);
        v8bf hi = *(const v8bf*)(ap + k0 + 16);
        v16bf a = __builtin_shufflevector(lo, hi, 0, 1, 2, 3, 4, 5, 6, 7,
                                                  8, 9, 10, 11, 12, 13, 14, 15);
        v16bf b[4];
#pragma unroll
        for (int t = 0; t < 4; ++t) {
            const __bf16* bl = &ldsB[(t * 16 + l16) * KP + half * 8 + k0];
            v8bf blo = *(const v8bf*)bl;            // ds_load_b128 x2, no conflicts
            v8bf bhi = *(const v8bf*)(bl + 16);
            b[t] = __builtin_shufflevector(blo, bhi, 0, 1, 2, 3, 4, 5, 6, 7,
                                                     8, 9, 10, 11, 12, 13, 14, 15);
        }
        acc0 = __builtin_amdgcn_wmma_f32_16x16x32_bf16(false, a, false, b[0], (short)0, acc0, false, false);
        acc1 = __builtin_amdgcn_wmma_f32_16x16x32_bf16(false, a, false, b[1], (short)0, acc1, false, false);
        acc2 = __builtin_amdgcn_wmma_f32_16x16x32_bf16(false, a, false, b[2], (short)0, acc2, false, false);
        acc3 = __builtin_amdgcn_wmma_f32_16x16x32_bf16(false, a, false, b[3], (short)0, acc3, false, false);
    }

    // D layout: vgpr r -> row r (lanes 0-15) / r+8 (lanes 16-31), lane = column.
#pragma unroll
    for (int t = 0; t < 4; ++t) {
        const v8f acc = (t == 0) ? acc0 : (t == 1) ? acc1 : (t == 2) ? acc2 : acc3;
        const int col = colBase + t * 16 + l16;
        const float bv = bias ? bias[col] : 0.0f;
#pragma unroll
        for (int r = 0; r < 8; ++r) {
            const int row = rowBase + half * 8 + r;
            const float y = acc[r] + bv;
            if (BF16OUT)
                ((__bf16*)Yv)[(size_t)row * m + col] = f2bf(y);
            else
                ((float*)Yv)[(size_t)row * m + col] = y;
        }
    }
}

// ---------------------------------------------------------------------------
// Pass A: per (edge, head) attention logit + segment max over dst.
// One wave per (e,h); 128-wide dot product, lanes stride by 32.
// ---------------------------------------------------------------------------
__global__ __launch_bounds__(256)
void attn_logits(const float* __restrict__ q, const float* __restrict__ kk,
                 const __bf16* __restrict__ ebuf, const int* __restrict__ eidx,
                 float* __restrict__ abuf, unsigned* __restrict__ amax,
                 int EH, float rscale)
{
    const int gid  = blockIdx.x * blockDim.x + threadIdx.x;
    const int wid  = gid >> 5;
    const int lane = gid & 31;
    if (wid >= EH) return;
    const int e = wid >> 3, h = wid & 7;
    const int src = eidx[2 * e + 0];
    const int dst = eidx[2 * e + 1];

    const float*  qp = q    + (size_t)dst * HC + h * 128;
    const float*  kp = kk   + (size_t)src * HC + h * 128;
    const __bf16* ep = ebuf + (size_t)e   * HC + h * 128;

    float s = 0.0f;
#pragma unroll
    for (int c = lane; c < 128; c += 32)
        s += qp[c] * (kp[c] + bf2f(ep[c]));
#pragma unroll
    for (int off = 16; off > 0; off >>= 1)
        s += __shfl_xor(s, off, 32);

    if (lane == 0) {
        s *= rscale;
        abuf[wid] = s;
        atomicMax(&amax[(size_t)dst * HH + h], flipf(s));
    }
}

// ---------------------------------------------------------------------------
// Pass B: ex = exp(a - amax[dst]); segment-sum denominator. In-place on abuf.
// ---------------------------------------------------------------------------
__global__ __launch_bounds__(256)
void attn_exp(const int* __restrict__ eidx, float* __restrict__ abuf,
              const unsigned* __restrict__ amax, float* __restrict__ den, int EH)
{
    const int i = blockIdx.x * blockDim.x + threadIdx.x;
    if (i >= EH) return;
    const int e = i >> 3, h = i & 7;
    const int dst = eidx[2 * e + 1];
    const float m  = unflipf(amax[(size_t)dst * HH + h]);
    const float ex = __expf(abuf[i] - m);
    abuf[i] = ex;
    atomicAdd(&den[(size_t)dst * HH + h], ex);
}

// ---------------------------------------------------------------------------
// Pass C: agg[dst] += mean_h( alpha_h * (v[src]+e) ).  128 threads = channels.
// ---------------------------------------------------------------------------
__global__ __launch_bounds__(128)
void attn_aggregate(const float* __restrict__ v, const __bf16* __restrict__ ebuf,
                    const int* __restrict__ eidx, const float* __restrict__ abuf,
                    const float* __restrict__ den, float* __restrict__ agg)
{
    const int e = blockIdx.x;
    const int c = threadIdx.x;
    const int src = eidx[2 * e + 0];
    const int dst = eidx[2 * e + 1];
    float acc = 0.0f;
#pragma unroll
    for (int h = 0; h < HH; ++h) {
        const float alpha = abuf[(size_t)e * HH + h] / den[(size_t)dst * HH + h];
        acc += alpha * (v[(size_t)src * HC + h * 128 + c] +
                        bf2f(ebuf[(size_t)e * HC + h * 128 + c]));
    }
    atomicAdd(&agg[(size_t)dst * 128 + c], acc * 0.125f);
}

// ---------------------------------------------------------------------------
// out = tanh(GroupNorm(a + b)) ; C=128, 16 groups of 8.
// ---------------------------------------------------------------------------
__global__ __launch_bounds__(128)
void add_gn_tanh(const float* __restrict__ a, const float* __restrict__ b,
                 const float* __restrict__ w, const float* __restrict__ bi,
                 float* __restrict__ out)
{
    const int n = blockIdx.x;
    const int c = threadIdx.x;
    __shared__ float xs[128];
    __shared__ float mu[GRP], rs[GRP];

    const float x = a[(size_t)n * 128 + c] + b[(size_t)n * 128 + c];
    xs[c] = x;
    __syncthreads();
    if (c < GRP) {
        float m = 0.0f;
#pragma unroll
        for (int j = 0; j < 8; ++j) m += xs[c * 8 + j];
        m *= 0.125f;
        float vv = 0.0f;
#pragma unroll
        for (int j = 0; j < 8; ++j) { float d = xs[c * 8 + j] - m; vv += d * d; }
        vv *= 0.125f;
        mu[c] = m;
        rs[c] = rsqrtf(vv + EPSV);
    }
    __syncthreads();
    const int g = c >> 3;
    out[(size_t)n * 128 + c] = tanhf((x - mu[g]) * rs[g] * w[c] + bi[c]);
}

// ---------------------------------------------------------------------------
// host side
// ---------------------------------------------------------------------------
template <int K, bool BF16OUT>
static inline void gemm_launch(const __bf16* Xb, const __bf16* Wt, const float* bias,
                               void* Y, int n, int m, hipStream_t s)
{
    dim3 grid(m / 64, (n + 63) / 64);
    wmma_gemm_bf16<K, BF16OUT><<<grid, 128, 0, s>>>(Xb, Wt, bias, Y, n, m);
}

static inline void cvt_launch(const float* in, __bf16* out, int n, hipStream_t s)
{
    cvt_f32_bf16<<<(n + 255) / 256, 256, 0, s>>>(in, out, n);
}

static inline void cvtT_launch(const float* W, __bf16* Wt, int k, int m, hipStream_t s)
{
    cvt_transpose_f32_bf16<<<(k * m + 255) / 256, 256, 0, s>>>(W, Wt, k, m);
}

extern "C" void kernel_launch(void* const* d_in, const int* in_sizes, int n_in,
                              void* d_out, int out_size, void* d_ws, size_t ws_size,
                              hipStream_t stream)
{
    // input order: x, edge_index, edge_attr, then params in dict order
    const float* x    = (const float*)d_in[0];
    const int*   eidx = (const int*)  d_in[1];     // (E,2): src, dst
    const float* ea   = (const float*)d_in[2];
    const float *Wq1 = (const float*)d_in[3],  *bq1 = (const float*)d_in[4];
    const float *Wk1 = (const float*)d_in[5],  *bk1 = (const float*)d_in[6];
    const float *Wv1 = (const float*)d_in[7],  *bv1 = (const float*)d_in[8];
    const float *We1 = (const float*)d_in[9];
    const float *Ws1 = (const float*)d_in[10], *bs1 = (const float*)d_in[11];
    const float *gw1 = (const float*)d_in[12], *gb1 = (const float*)d_in[13];
    const float *Wq2 = (const float*)d_in[14], *bq2 = (const float*)d_in[15];
    const float *Wk2 = (const float*)d_in[16], *bk2 = (const float*)d_in[17];
    const float *Wv2 = (const float*)d_in[18], *bv2 = (const float*)d_in[19];
    const float *We2 = (const float*)d_in[20];
    const float *Ws2 = (const float*)d_in[21], *bs2 = (const float*)d_in[22];
    const float *gw2 = (const float*)d_in[23], *gb2 = (const float*)d_in[24];

    // workspace carve-out
    char* p = (char*)d_ws;
    auto take = [&](size_t bytes) {
        void* r = (void*)p;
        p += (bytes + 255) & ~(size_t)255;
        return r;
    };
    float*    qb   = (float*)   take((size_t)NN * HC * 4);       //  41 MB
    float*    kb   = (float*)   take((size_t)NN * HC * 4);       //  41 MB
    float*    vb   = (float*)   take((size_t)NN * HC * 4);       //  41 MB
    __bf16*   eb   = (__bf16*)  take((size_t)EE * HC * 2);       // 328 MB
    float*    skip = (float*)   take((size_t)NN * 128 * 4);
    float*    h1   = (float*)   take((size_t)NN * 128 * 4);
    float*    ab   = (float*)   take((size_t)EE * HH * 4);
    unsigned* amax = (unsigned*)take((size_t)NN * HH * 4);
    float*    den  = (float*)   take((size_t)NN * HH * 4);
    float*    agg  = (float*)   take((size_t)NN * 128 * 4);
    __bf16*   xb   = (__bf16*)  take((size_t)NN * INC * 2);      // bf16 x
    __bf16*   eab  = (__bf16*)  take((size_t)EE * EDC * 2);      // bf16 edge_attr
    __bf16*   h1b  = (__bf16*)  take((size_t)NN * MIDC * 2);     // bf16 h1
    __bf16*   wt   = (__bf16*)  take((size_t)INC * HC * 2);      // transposed weight

    const float rscale   = 0.08838834764831845f;   // 1/sqrt(128)
    const int   EH       = EE * HH;
    const int   logitBlk = (EH * 32) / 256;        // exact: E*H*32 % 256 == 0
    const int   expBlk   = (EH + 255) / 256;

    // ============================= layer 1 =============================
    cvt_launch(x,  xb,  NN * INC, stream);
    cvt_launch(ea, eab, EE * EDC, stream);

    cvtT_launch(Wq1, wt, INC, HC, stream);
    gemm_launch<INC, false>(xb, wt, bq1, qb, NN, HC, stream);
    cvtT_launch(Wk1, wt, INC, HC, stream);
    gemm_launch<INC, false>(xb, wt, bk1, kb, NN, HC, stream);
    cvtT_launch(Wv1, wt, INC, HC, stream);
    gemm_launch<INC, false>(xb, wt, bv1, vb, NN, HC, stream);
    cvtT_launch(We1, wt, EDC, HC, stream);
    gemm_launch<EDC, true>(eab, wt, nullptr, eb, EE, HC, stream);
    cvtT_launch(Ws1, wt, INC, 128, stream);
    gemm_launch<INC, false>(xb, wt, bs1, skip, NN, 128, stream);

    (void)hipMemsetAsync(amax, 0, (size_t)NN * HH * 4, stream);
    (void)hipMemsetAsync(den,  0, (size_t)NN * HH * 4, stream);
    (void)hipMemsetAsync(agg,  0, (size_t)NN * 128 * 4, stream);

    attn_logits<<<logitBlk, 256, 0, stream>>>(qb, kb, eb, eidx, ab, amax, EH, rscale);
    attn_exp<<<expBlk, 256, 0, stream>>>(eidx, ab, amax, den, EH);
    attn_aggregate<<<EE, 128, 0, stream>>>(vb, eb, eidx, ab, den, agg);
    add_gn_tanh<<<NN, 128, 0, stream>>>(agg, skip, gw1, gb1, h1);

    // ============================= layer 2 =============================
    cvt_launch(h1, h1b, NN * MIDC, stream);

    cvtT_launch(Wq2, wt, MIDC, HC, stream);
    gemm_launch<MIDC, false>(h1b, wt, bq2, qb, NN, HC, stream);
    cvtT_launch(Wk2, wt, MIDC, HC, stream);
    gemm_launch<MIDC, false>(h1b, wt, bk2, kb, NN, HC, stream);
    cvtT_launch(Wv2, wt, MIDC, HC, stream);
    gemm_launch<MIDC, false>(h1b, wt, bv2, vb, NN, HC, stream);
    cvtT_launch(We2, wt, EDC, HC, stream);
    gemm_launch<EDC, true>(eab, wt, nullptr, eb, EE, HC, stream);
    cvtT_launch(Ws2, wt, MIDC, 128, stream);
    gemm_launch<MIDC, false>(h1b, wt, bs2, skip, NN, 128, stream);

    (void)hipMemsetAsync(amax, 0, (size_t)NN * HH * 4, stream);
    (void)hipMemsetAsync(den,  0, (size_t)NN * HH * 4, stream);
    (void)hipMemsetAsync(agg,  0, (size_t)NN * 128 * 4, stream);

    attn_logits<<<logitBlk, 256, 0, stream>>>(qb, kb, eb, eidx, ab, amax, EH, rscale);
    attn_exp<<<expBlk, 256, 0, stream>>>(eidx, ab, amax, den, EH);
    attn_aggregate<<<EE, 128, 0, stream>>>(vb, eb, eidx, ab, den, agg);
    add_gn_tanh<<<NN, 128, 0, stream>>>(agg, skip, gw2, gb2, (float*)d_out);
}
